// ParallelUnifiedAttention_33260226740873
// MI455X (gfx1250) — matrix operand order repaired and checked by reading, not compiled, not run on hardware
//
#include <hip/hip_runtime.h>
#include <hip/hip_bf16.h>
#include <math.h>

typedef _Float16 half_t;
typedef __attribute__((ext_vector_type(16))) _Float16 v16h;
typedef __attribute__((ext_vector_type(8)))  _Float16 v8h;
typedef __attribute__((ext_vector_type(4)))  _Float16 v4h;
typedef __attribute__((ext_vector_type(8)))  float    v8f;
typedef __attribute__((ext_vector_type(4)))  float    v4f;

#define S_DIM 4096
#define B_DIM 8
#define E_DIM 1024
#define H_DIM 8
#define F_DIM 8192          // E*H

#define FTILE 128           // f-columns per workgroup (8 waves x 16)
#define SCHUNK 32           // s-rows per chunk (two WMMA M-tiles)
#define NCHUNKS (S_DIM / SCHUNK)   // 128
#define KSTEP 32            // WMMA K
#define NKSTEPS (E_DIM / KSTEP)    // 32
#define BLOCK 256           // 8 waves

// Unpadded rows (2048 B stride) + XOR swizzle for bank-conflict-free column
// access:  byte_addr = row*2048 + (kbyte ^ ((row & 15) << 4)).
// XOR operand is a multiple of 16 -> every 16B-aligned access stays 16B-aligned
// and bytes within a 16B block stay together.  Banks for 16 consecutive rows at
// equal kbyte: ((kb>>2) ^ (4*row)) mod 64 -> 16 distinct banks.
#define ROWB 2048
#define WQ_BYTES (FTILE * ROWB)             // 256 KB
#define X_BYTES  (SCHUNK * ROWB)            // 64 KB
#define SMEM_BYTES ((size_t)(WQ_BYTES + X_BYTES))   // 320 KB WGP LDS, exactly

__device__ __forceinline__ int swz(int row, int kbyte) {
  return row * ROWB + (kbyte ^ ((row & 15) << 4));
}

__device__ __forceinline__ v16h cat8(v8h a, v8h b) {
  return __builtin_shufflevector(a, b, 0,1,2,3,4,5,6,7,8,9,10,11,12,13,14,15);
}

__device__ __forceinline__ v4h cvt4(v4f v) {
  v4h h;
  h[0] = (half_t)v[0]; h[1] = (half_t)v[1];
  h[2] = (half_t)v[2]; h[3] = (half_t)v[3];
  return h;
}

// ---------------------------------------------------------------------------
// Kernel 1: fused scores GEMM + online softmax over S + weighted x reduction.
//   head_out[b, f] = sum_s softmax_s(x[s,b,:]·Wq[f,:]) * x[s, b, f % E]
// (bq[f] is constant over s, so it cancels in the softmax and is skipped.)
// Each wave: one 16-col N tile, two 16-row M tiles per chunk -> each B (Wq)
// fetch feeds two WMMAs (3 ds_load_b128 per WMMA instead of 4).
// ---------------------------------------------------------------------------
__global__ void __launch_bounds__(BLOCK, 1)
attn_fused_kernel(const float* __restrict__ x,
                  const float* __restrict__ Wq,
                  float* __restrict__ head_out)
{
  extern __shared__ char smem[];
  char* __restrict__ wqs = smem;              // [FTILE][2048 B], swizzled
  char* __restrict__ xs  = smem + WQ_BYTES;   // [SCHUNK][2048 B], swizzled

  const int b    = blockIdx.x & (B_DIM - 1);
  const int tile = blockIdx.x >> 3;          // 0..63
  const int f0   = tile * FTILE;
  const int e0   = f0 & (E_DIM - 1);         // tile never crosses an h boundary

  const int t    = threadIdx.x;
  const int wave = t >> 5;
  const int lane = t & 31;
  const int ln   = lane & 15;                // column within the wave's 16x16 tile
  const int hs   = lane >> 4;                // half-select (row group / K group)

  // ---- load Wq tile -> LDS as f16 (2 threads per row, interleaved v4) ----
  {
    const int fr = t >> 1;
    const int q  = t & 1;
    const float* src = Wq + (size_t)(f0 + fr) * E_DIM;
    #pragma unroll 8
    for (int j = 0; j < 128; ++j) {
      const int c = q * 4 + j * 8;
      v4f v = *(const v4f*)(src + c);
      *(v4h*)(wqs + swz(fr, c * 2)) = cvt4(v);
    }
  }

  // ---- x-chunk register pipeline: 8 threads/row, 128 floats per thread ----
  const int xr = t >> 3;             // row 0..31 within chunk
  const int xp = t & 7;              // position within row
  v4f xb[32];

  // prefetch chunk 0 (lane-interleaved for coalescing)
  {
    const float* src = x + ((size_t)xr * B_DIM + b) * E_DIM;
    #pragma unroll
    for (int j = 0; j < 32; ++j) xb[j] = *(const v4f*)(src + xp * 4 + j * 32);
  }

  // per-lane online-softmax state for column (f0 + wave*16 + ln)
  float m   = -INFINITY;
  float Z   = 0.0f;
  float num = 0.0f;
  const int fcol = f0 + wave * 16 + ln;
  const int ecb  = (e0 + wave * 16 + ln) * 2;   // byte offset of this column in x rows

  const char* __restrict__ xa0 = xs + (ln)      * ROWB;      // A row (M tile 0)
  const char* __restrict__ xa1 = xs + (16 + ln) * ROWB;      // A row (M tile 1)
  const char* __restrict__ wqb = wqs + (wave * 16 + ln) * ROWB;  // B column
  const int asw = ln << 4;   // swizzle term: (row & 15) << 4 == ln<<4 for all three

  for (int sc = 0; sc < NCHUNKS; ++sc) {
    // commit prefetched x chunk to LDS (f16, swizzled)
    #pragma unroll
    for (int j = 0; j < 32; ++j) {
      const int c = xp * 4 + j * 32;
      *(v4h*)(xs + swz(xr, c * 2)) = cvt4(xb[j]);
    }
    __syncthreads();

    // issue global loads for the next chunk (overlaps with WMMA below)
    if (sc + 1 < NCHUNKS) {
      const float* src = x + ((size_t)((sc + 1) * SCHUNK + xr) * B_DIM + b) * E_DIM;
      #pragma unroll
      for (int j = 0; j < 32; ++j) xb[j] = *(const v4f*)(src + xp * 4 + j * 32);
    }

    // ---- 32x16 score tile: 32 k-steps, B fetched once per two WMMAs ----
    v8f c0 = {};   // rows 0..15 of chunk
    v8f c1 = {};   // rows 16..31 of chunk
    #pragma unroll 4
    for (int kk = 0; kk < NKSTEPS; ++kk) {
      const int kb = kk * (KSTEP * 2);   // k-step base in bytes
      // A (x), 16-bit A layout: lane<16 -> K {0..7,16..23}, lane>=16 -> +8
      v8h a00 = *(const v8h*)(xa0 + ((kb + hs * 16)      ^ asw));
      v8h a01 = *(const v8h*)(xa0 + ((kb + 32 + hs * 16) ^ asw));
      v8h a10 = *(const v8h*)(xa1 + ((kb + hs * 16)      ^ asw));
      v8h a11 = *(const v8h*)(xa1 + ((kb + 32 + hs * 16) ^ asw));
      // B (Wq), same K order as A: lane<16 -> K {0..7,16..23}, lane>=16 -> +8
      v8h b0  = *(const v8h*)(wqb + ((kb + hs * 16)      ^ asw));
      v8h b1  = *(const v8h*)(wqb + ((kb + 32 + hs * 16) ^ asw));
      v16h A0 = cat8(a00, a01);
      v16h A1 = cat8(a10, a11);
      v16h Bm = cat8(b0, b1);
      c0 = __builtin_amdgcn_wmma_f32_16x16x32_f16(false, A0, false, Bm,
                                                  (short)0, c0, false, false);
      c1 = __builtin_amdgcn_wmma_f32_16x16x32_f16(false, A1, false, Bm,
                                                  (short)0, c1, false, false);
    }

    // ---- online softmax update (column resides in lanes L and L^16) ----
    float cmax = -INFINITY;
    #pragma unroll
    for (int i = 0; i < 8; ++i) {
      cmax = fmaxf(cmax, fmaxf(c0[i], c1[i]));
    }
    cmax = fmaxf(cmax, __shfl_xor(cmax, 16));
    const float newm = fmaxf(m, cmax);

    float zc = 0.0f, nc = 0.0f;
    #pragma unroll
    for (int i = 0; i < 8; ++i) {
      const int r0 = hs * 8 + i;        // rows of M tile 0
      const int r1 = 16 + hs * 8 + i;   // rows of M tile 1
      float w0  = __expf(c0[i] - newm);
      float w1  = __expf(c1[i] - newm);
      float xv0 = (float)*(const half_t*)(xs + swz(r0, ecb));
      float xv1 = (float)*(const half_t*)(xs + swz(r1, ecb));
      zc += w0 + w1;
      nc += w0 * xv0 + w1 * xv1;
    }
    zc += __shfl_xor(zc, 16);
    nc += __shfl_xor(nc, 16);

    const float scale = __expf(m - newm);   // exp(-inf)=0 handles the first chunk
    Z   = Z   * scale + zc;
    num = num * scale + nc;
    m   = newm;

    __syncthreads();   // protect xs before next chunk's commit
  }

  if (hs == 0)
    head_out[(size_t)b * F_DIM + fcol] = num / Z;
}

// ---------------------------------------------------------------------------
// Kernel 2: out[b, j] = head_out[b, :] · Wc[j, :] + bc[j]
// One wave per output element; coalesced float4 reads of Wc (memory-bound).
// ---------------------------------------------------------------------------
__global__ void __launch_bounds__(BLOCK, 1)
combine_kernel(const float* __restrict__ head_out,
               const float* __restrict__ Wc,
               const float* __restrict__ bc,
               float* __restrict__ out)
{
  const int b    = blockIdx.y;
  const int wave = threadIdx.x >> 5;
  const int lane = threadIdx.x & 31;
  const int j    = blockIdx.x * 8 + wave;   // gridDim.x = 128 -> j in [0,1024)

  const float* __restrict__ wrow = Wc + (size_t)j * F_DIM;
  const float* __restrict__ hrow = head_out + (size_t)b * F_DIM;

  v4f acc = {0.f, 0.f, 0.f, 0.f};
  #pragma unroll 4
  for (int f = lane * 4; f < F_DIM; f += 32 * 4) {
    v4f wv = *(const v4f*)(wrow + f);
    v4f hv = *(const v4f*)(hrow + f);
    acc += wv * hv;
  }
  float s = acc[0] + acc[1] + acc[2] + acc[3];
  s += __shfl_xor(s, 16);
  s += __shfl_xor(s, 8);
  s += __shfl_xor(s, 4);
  s += __shfl_xor(s, 2);
  s += __shfl_xor(s, 1);
  if (lane == 0)
    out[(size_t)b * E_DIM + j] = s + bc[j];
}

// ---------------------------------------------------------------------------
extern "C" void kernel_launch(void* const* d_in, const int* in_sizes, int n_in,
                              void* d_out, int out_size, void* d_ws, size_t ws_size,
                              hipStream_t stream) {
  const float* x  = (const float*)d_in[0];   // (S, B, E)
  const float* Wq = (const float*)d_in[1];   // (E*H, E)
  // d_in[2] = bq : constant over S -> cancels in softmax, unused
  const float* Wc = (const float*)d_in[3];   // (E, E*H)
  const float* bc = (const float*)d_in[4];   // (E,)
  float* out = (float*)d_out;                // (B, E)

  float* head_out = (float*)d_ws;            // (B, F) = 256 KB scratch

  // 64 f-tiles x 8 batches = 512 workgroups of 8 waves
  attn_fused_kernel<<<dim3(512), dim3(BLOCK), SMEM_BYTES, stream>>>(x, Wq, head_out);

  combine_kernel<<<dim3(128, 8), dim3(BLOCK), 0, stream>>>(head_out, Wc, bc, out);
}
